// DeepNet_38491496907328
// MI455X (gfx1250) — matrix operand order repaired and checked by reading, not compiled, not run on hardware
//
#include <hip/hip_runtime.h>
#include <hip/hip_bf16.h>

// ---------------------------------------------------------------------------
// DeepNet on MI455X (gfx1250): bf16 WMMA GEMMs + f32 BN/residual epilogues.
// ---------------------------------------------------------------------------

typedef __attribute__((ext_vector_type(16))) __bf16 v16bf;
typedef __attribute__((ext_vector_type(8)))  __bf16 v8bf;
typedef __attribute__((ext_vector_type(8)))  float  v8f;
typedef __attribute__((ext_vector_type(4)))  int    v4i;

// Pointer-to-vector types in explicit address spaces for the async builtin:
// param 1 is '<4 x i32> addrspace(1)*' (clang prints AS1 as __device__).
typedef __attribute__((address_space(1))) v4i* as1_v4i_p;
typedef __attribute__((address_space(3))) v4i* as3_v4i_p;

#define N_ROWS 8192
#define H_DIM  4096
#define DEPTH  3
#define EPS    1e-5f

// GEMM tiling: 128x128 block tile, K-step 64, 8 wave32s (2x4), each wave 64x32.
#define TM   128
#define TN   128
#define TK   64
#define LDSS 72   // padded LDS row stride in bf16 elems (64 + 8): 144B rows, 16B aligned, conflict-spread

// gfx1250 async global->LDS path (builtin confirmed present by round-1 probe).
#ifndef HAVE_ASYNC
#if defined(__has_builtin)
#if __has_builtin(__builtin_amdgcn_global_load_async_to_lds_b128)
#define HAVE_ASYNC 1
#endif
#endif
#endif
#ifndef HAVE_ASYNC
#define HAVE_ASYNC 0
#endif

__device__ __forceinline__ void async_fence() {
#if HAVE_ASYNC
#if defined(__has_builtin) && __has_builtin(__builtin_amdgcn_s_wait_asynccnt)
  __builtin_amdgcn_s_wait_asynccnt(0);
#else
  asm volatile("s_wait_asynccnt 0" ::: "memory");
#endif
#endif
}

// Stage a 128-row x 64-col bf16 tile (row-major, leading dim ld) into LDS with
// padded stride LDSS. 256 threads * 4 chunks of 16B each.
__device__ __forceinline__ void stage_tile(const __bf16* __restrict__ g, int ld,
                                           __bf16* s, int tid) {
#pragma unroll
  for (int c = 0; c < 4; ++c) {
    int idx = tid + c * 256;
    int row = idx >> 3;
    int col = (idx & 7) * 8;
    const __bf16* gp = g + (size_t)row * ld + col;
    __bf16* sp = s + row * LDSS + col;
#if HAVE_ASYNC
    __builtin_amdgcn_global_load_async_to_lds_b128(
        (as1_v4i_p)(void*)gp, (as3_v4i_p)(void*)sp, 0, 0);
#else
    uint4 t = *(const uint4*)gp;
    *(uint4*)sp = t;
#endif
  }
}

// C[M,N] = relu(A[M,K] @ B[K,N] + bias[N]), A bf16 row-major, B given TRANSPOSED
// as Bt[N,K] bf16 row-major so both operand tiles stream contiguously in K.
__global__ __launch_bounds__(256)
void k_gemm_bias_relu(const __bf16* __restrict__ A,
                      const __bf16* __restrict__ Bt,
                      const float*  __restrict__ bias,
                      float*        __restrict__ C,
                      int Nc, int Kd) {
  __shared__ __attribute__((aligned(16))) __bf16 sA[2][TM * LDSS];
  __shared__ __attribute__((aligned(16))) __bf16 sB[2][TN * LDSS];

  const int tid  = threadIdx.x;
  const int lane = tid & 31;
  const int w    = tid >> 5;
  const int wm   = w >> 2;   // 0..1
  const int wn   = w & 3;    // 0..3

  const int m0 = blockIdx.y * TM;
  const int n0 = blockIdx.x * TN;

  const v8f vzero = {0.f, 0.f, 0.f, 0.f, 0.f, 0.f, 0.f, 0.f};
  v8f acc[4][2];
#pragma unroll
  for (int mt = 0; mt < 4; ++mt)
#pragma unroll
    for (int nt = 0; nt < 2; ++nt) acc[mt][nt] = vzero;

  stage_tile(A  + (size_t)m0 * Kd, Kd, sA[0], tid);
  stage_tile(Bt + (size_t)n0 * Kd, Kd, sB[0], tid);
  async_fence();
  __syncthreads();

  const int nK = Kd / TK;
  for (int ks = 0; ks < nK; ++ks) {
    const int cur = ks & 1, nxt = cur ^ 1;
    if (ks + 1 < nK) {
      stage_tile(A  + (size_t)m0 * Kd + (ks + 1) * TK, Kd, sA[nxt], tid);
      stage_tile(Bt + (size_t)n0 * Kd + (ks + 1) * TK, Kd, sB[nxt], tid);
    }
#pragma unroll
    for (int kk = 0; kk < TK; kk += 32) {
      // A fragment (16x32 bf16): lane L holds row M=L&15; lanes<16 carry
      // K {0..7,16..23}, lanes>=16 carry K {8..15,24..31}.
      v16bf af[4];
      const int aBase = (wm * 64 + (lane & 15)) * LDSS + kk + ((lane >> 4) << 3);
#pragma unroll
      for (int mt = 0; mt < 4; ++mt) {
        const __bf16* p = &sA[cur][aBase + mt * 16 * LDSS];
        v8bf lo = *(const v8bf*)p;
        v8bf hi = *(const v8bf*)(p + 16);
        af[mt] = __builtin_shufflevector(lo, hi, 0, 1, 2, 3, 4, 5, 6, 7,
                                         8, 9, 10, 11, 12, 13, 14, 15);
      }
      // B fragment (32x16 bf16): lane L holds col N=L&15; lanes<16 carry
      // K {0..7,16..23}, lanes>=16 carry K {8..15,24..31}.
      v16bf bfrag[2];
      const int bBase = (wn * 32 + (lane & 15)) * LDSS + kk + ((lane >> 4) << 3);
#pragma unroll
      for (int nt = 0; nt < 2; ++nt) {
        const __bf16* p = &sB[cur][bBase + nt * 16 * LDSS];
        v8bf lo = *(const v8bf*)p;
        v8bf hi = *(const v8bf*)(p + 16);
        bfrag[nt] = __builtin_shufflevector(lo, hi, 0, 1, 2, 3, 4, 5, 6, 7,
                                            8, 9, 10, 11, 12, 13, 14, 15);
      }
#pragma unroll
      for (int mt = 0; mt < 4; ++mt)
#pragma unroll
        for (int nt = 0; nt < 2; ++nt)
          acc[mt][nt] = __builtin_amdgcn_wmma_f32_16x16x32_bf16(
              false, af[mt], false, bfrag[nt], (short)0, acc[mt][nt],
              false, false);
    }
    async_fence();
    __syncthreads();
  }

  // Epilogue: D layout = lane L holds col N=L&15, rows v + 8*(L>>4).
  const int colBase = n0 + wn * 32 + (lane & 15);
  const int rowHalf = (lane >> 4) * 8;
#pragma unroll
  for (int mt = 0; mt < 4; ++mt) {
#pragma unroll
    for (int nt = 0; nt < 2; ++nt) {
      const int gc = colBase + nt * 16;
      const float bv = bias[gc];
#pragma unroll
      for (int v = 0; v < 8; ++v) {
        float x = acc[mt][nt][v] + bv;
        x = x > 0.f ? x : 0.f;
        const int gr = m0 + wm * 64 + mt * 16 + rowHalf + v;
        C[(size_t)gr * Nc + gc] = x;
      }
    }
  }
}

// First layer (K=2): X = relu(X_in @ W_first + b_first); also bf16 copy.
__global__ void k_first(const float* __restrict__ Xin, const float* __restrict__ Wf,
                        const float* __restrict__ bf0, float* __restrict__ X,
                        __bf16* __restrict__ Abf) {
  const int idx = blockIdx.x * 256 + threadIdx.x;
  const int h = idx & (H_DIM - 1);
  const int n = idx >> 12;
  float x = Xin[n * 2] * Wf[h] + Xin[n * 2 + 1] * Wf[H_DIM + h] + bf0[h];
  x = x > 0.f ? x : 0.f;
  X[idx] = x;
  Abf[idx] = (__bf16)x;
}

// Transpose W[k][n] (f32) -> Wt[n][k] (bf16) in 32x32 LDS tiles.
__global__ void k_transpose_bf16(const float* __restrict__ W, __bf16* __restrict__ Wt) {
  __shared__ float tile[32][33];
  const int bx = blockIdx.x * 32;  // n base
  const int by = blockIdx.y * 32;  // k base
  const int tx = threadIdx.x & 31;
  const int ty = threadIdx.x >> 5;  // 0..7
  for (int r = ty; r < 32; r += 8)
    tile[r][tx] = W[(size_t)(by + r) * H_DIM + bx + tx];
  __syncthreads();
  for (int r = ty; r < 32; r += 8)
    Wt[(size_t)(bx + r) * H_DIM + by + tx] = (__bf16)tile[tx][r];
}

// rowSq[n] = sum_h Xh[n,h]^2 (deterministic tree reduction, one block per row).
__global__ void k_rowsq(const float* __restrict__ Xh, float* __restrict__ rowSq) {
  __shared__ float red[256];
  const int n = blockIdx.x;
  const float* row = Xh + (size_t)n * H_DIM;
  float s = 0.f;
  for (int h = threadIdx.x; h < H_DIM; h += 256) { float v = row[h]; s += v * v; }
  red[threadIdx.x] = s;
  __syncthreads();
  for (int off = 128; off > 0; off >>= 1) {
    if (threadIdx.x < off) red[threadIdx.x] += red[threadIdx.x + off];
    __syncthreads();
  }
  if (threadIdx.x == 0) rowSq[n] = red[0];
}

#define RBLK 32  // row-block partials for deterministic column reduction

// Y = rowSq + Xh + X (overwrites Xh); per-row-block column partial sums.
__global__ void k_combine_colpart(const float* __restrict__ X, float* __restrict__ Xh,
                                  const float* __restrict__ rowSq,
                                  float* __restrict__ colP, float* __restrict__ colPS) {
  const int h  = blockIdx.x * 256 + threadIdx.x;
  const int n0 = blockIdx.y * (N_ROWS / RBLK);
  float s = 0.f, s2 = 0.f;
  for (int r = 0; r < N_ROWS / RBLK; ++r) {
    const int n = n0 + r;
    const size_t idx = (size_t)n * H_DIM + h;
    const float y = rowSq[n] + Xh[idx] + X[idx];
    Xh[idx] = y;
    s += y;
    s2 += y * y;
  }
  colP [(size_t)blockIdx.y * H_DIM + h] = s;
  colPS[(size_t)blockIdx.y * H_DIM + h] = s2;
}

__global__ void k_colreduce(const float* __restrict__ colP, const float* __restrict__ colPS,
                            float* __restrict__ colMu, float* __restrict__ colIv) {
  const int h = blockIdx.x * 256 + threadIdx.x;
  float s = 0.f, s2 = 0.f;
  for (int b = 0; b < RBLK; ++b) {
    s  += colP [(size_t)b * H_DIM + h];
    s2 += colPS[(size_t)b * H_DIM + h];
  }
  const float mu  = s * (1.0f / N_ROWS);
  const float var = s2 * (1.0f / N_ROWS) - mu * mu;
  colMu[h] = mu;
  colIv[h] = rsqrtf(var + EPS);
}

__global__ void k_normalize(const float* __restrict__ Y, const float* __restrict__ colMu,
                            const float* __restrict__ colIv, float* __restrict__ X,
                            __bf16* __restrict__ Abf) {
  const int idx = blockIdx.x * 256 + threadIdx.x;
  const int h = idx & (H_DIM - 1);
  const float v = (Y[idx] - colMu[h]) * colIv[h];
  X[idx] = v;
  Abf[idx] = (__bf16)v;
}

// Final layer: Z[n,k] = X[n,:]·W_last[:,k] + b_last[k] + X_in[n,k] (wave per row).
__global__ void k_last_mm(const float* __restrict__ X, const float* __restrict__ Wl,
                          const float* __restrict__ bl, const float* __restrict__ Xin,
                          float* __restrict__ Z) {
  const int row  = (blockIdx.x * blockDim.x + threadIdx.x) >> 5;
  const int lane = threadIdx.x & 31;
  const float* xr = X + (size_t)row * H_DIM;
  float p0 = 0.f, p1 = 0.f;
  for (int j = lane; j < H_DIM; j += 32) {
    const float x = xr[j];
    p0 += x * Wl[j * 2 + 0];
    p1 += x * Wl[j * 2 + 1];
  }
#pragma unroll
  for (int off = 16; off > 0; off >>= 1) {
    p0 += __shfl_xor(p0, off, 32);
    p1 += __shfl_xor(p1, off, 32);
  }
  if (lane == 0) {
    Z[row * 2 + 0] = p0 + bl[0] + Xin[row * 2 + 0];
    Z[row * 2 + 1] = p1 + bl[1] + Xin[row * 2 + 1];
  }
}

__global__ void k_last_stats(const float* __restrict__ Z, float* __restrict__ st) {
  __shared__ float r0[256], r1[256], q0[256], q1[256];
  float s0 = 0.f, s1 = 0.f, t0 = 0.f, t1 = 0.f;
  for (int n = threadIdx.x; n < N_ROWS; n += 256) {
    const float a = Z[n * 2], b = Z[n * 2 + 1];
    s0 += a; t0 += a * a; s1 += b; t1 += b * b;
  }
  r0[threadIdx.x] = s0; r1[threadIdx.x] = s1;
  q0[threadIdx.x] = t0; q1[threadIdx.x] = t1;
  __syncthreads();
  for (int off = 128; off > 0; off >>= 1) {
    if (threadIdx.x < off) {
      r0[threadIdx.x] += r0[threadIdx.x + off];
      r1[threadIdx.x] += r1[threadIdx.x + off];
      q0[threadIdx.x] += q0[threadIdx.x + off];
      q1[threadIdx.x] += q1[threadIdx.x + off];
    }
    __syncthreads();
  }
  if (threadIdx.x == 0) {
    const float mu0 = r0[0] * (1.0f / N_ROWS);
    const float mu1 = r1[0] * (1.0f / N_ROWS);
    st[0] = mu0; st[1] = rsqrtf(q0[0] * (1.0f / N_ROWS) - mu0 * mu0 + EPS);
    st[2] = mu1; st[3] = rsqrtf(q1[0] * (1.0f / N_ROWS) - mu1 * mu1 + EPS);
  }
}

__global__ void k_last_norm(const float* __restrict__ Z, const float* __restrict__ st,
                            float* __restrict__ out) {
  const int idx = blockIdx.x * 256 + threadIdx.x;
  const int k = idx & 1;
  out[idx] = (Z[idx] - st[2 * k]) * st[2 * k + 1];
}

extern "C" void kernel_launch(void* const* d_in, const int* in_sizes, int n_in,
                              void* d_out, int out_size, void* d_ws, size_t ws_size,
                              hipStream_t stream) {
  const float* Xin = (const float*)d_in[0];
  const float* Wf  = (const float*)d_in[1];
  const float* bf0 = (const float*)d_in[2];
  const float* Wh  = (const float*)d_in[3];
  const float* bh  = (const float*)d_in[4];
  const float* Wl  = (const float*)d_in[5];
  const float* bl  = (const float*)d_in[6];
  float* out = (float*)d_out;

  char* ws = (char*)d_ws;
  auto alloc = [&](size_t bytes) {
    char* p = ws;
    ws += (bytes + 255) & ~(size_t)255;
    return p;
  };
  float*  X     = (float*) alloc((size_t)N_ROWS * H_DIM * 4);
  float*  Xh    = (float*) alloc((size_t)N_ROWS * H_DIM * 4);
  __bf16* Abf   = (__bf16*)alloc((size_t)N_ROWS * H_DIM * 2);
  __bf16* Wt    = (__bf16*)alloc((size_t)H_DIM * H_DIM * 2);
  float*  rowSq = (float*) alloc((size_t)N_ROWS * 4);
  float*  colP  = (float*) alloc((size_t)RBLK * H_DIM * 4);
  float*  colPS = (float*) alloc((size_t)RBLK * H_DIM * 4);
  float*  colMu = (float*) alloc((size_t)H_DIM * 4);
  float*  colIv = (float*) alloc((size_t)H_DIM * 4);
  float*  Z     = (float*) alloc((size_t)N_ROWS * 2 * 4);
  float*  st    = (float*) alloc(256);

  const dim3 b256(256);

  k_first<<<dim3((N_ROWS * H_DIM) / 256), b256, 0, stream>>>(Xin, Wf, bf0, X, Abf);

  for (int i = 0; i < DEPTH; ++i) {
    k_transpose_bf16<<<dim3(H_DIM / 32, H_DIM / 32), b256, 0, stream>>>(
        Wh + (size_t)i * H_DIM * H_DIM, Wt);
    k_gemm_bias_relu<<<dim3(H_DIM / TN, N_ROWS / TM), b256, 0, stream>>>(
        Abf, Wt, bh + (size_t)i * H_DIM, Xh, H_DIM, H_DIM);
    k_rowsq<<<dim3(N_ROWS), b256, 0, stream>>>(Xh, rowSq);
    k_combine_colpart<<<dim3(H_DIM / 256, RBLK), b256, 0, stream>>>(X, Xh, rowSq, colP, colPS);
    k_colreduce<<<dim3(H_DIM / 256), b256, 0, stream>>>(colP, colPS, colMu, colIv);
    k_normalize<<<dim3((N_ROWS * H_DIM) / 256), b256, 0, stream>>>(Xh, colMu, colIv, X, Abf);
  }

  k_last_mm<<<dim3(N_ROWS / 8), b256, 0, stream>>>(X, Wl, bl, Xin, Z);
  k_last_stats<<<dim3(1), b256, 0, stream>>>(Z, st);
  k_last_norm<<<dim3((N_ROWS * 2) / 256), b256, 0, stream>>>(Z, st, out);
}
